// Attention2D_23347442221284
// MI455X (gfx1250) — matrix operand order repaired and checked by reading, not compiled, not run on hardware
//
#include <hip/hip_runtime.h>

typedef __bf16 bf16_t;
typedef __attribute__((ext_vector_type(16))) __bf16 v16bf;
typedef __attribute__((ext_vector_type(8)))  float  v8f;
typedef __attribute__((ext_vector_type(4)))  unsigned int v4u;

union Frag16 { v16bf v; v4u q[2]; };

__device__ __forceinline__ v8f wmma_bf16(v16bf a, v16bf b, v8f c) {
  // D(16x16 f32) = A(16x32 bf16) * B(32x16 bf16) + C
  return __builtin_amdgcn_wmma_f32_16x16x32_bf16(false, a, false, b, (short)0, c, false, false);
}

// A-fragment: row per lane (ln); 16 contiguous K values starting at lg*16 (arow pre-offset)
__device__ __forceinline__ void loadA(Frag16& f, const bf16_t* arow, int k0) {
  f.q[0] = *reinterpret_cast<const v4u*>(arow + k0);
  f.q[1] = *reinterpret_cast<const v4u*>(arow + k0 + 8);
}
// B-fragment: col per lane; 16 contiguous K values starting at lg*16 (bcol pre-offset)
__device__ __forceinline__ void loadB(Frag16& f, const bf16_t* bcol, int k0) {
  f.q[0] = *reinterpret_cast<const v4u*>(bcol + k0);
  f.q[1] = *reinterpret_cast<const v4u*>(bcol + k0 + 8);
}

// ---------- weight convert + transpose: w[k][n] (512x512 f32) -> wt[n][k] bf16
__global__ __launch_bounds__(256) void wt_kernel(const float* __restrict__ w,
                                                 bf16_t* __restrict__ wt) {
  int idx = blockIdx.x * 256 + threadIdx.x;   // 0..262143
  int k = idx >> 9, n = idx & 511;
  wt[n * 512 + k] = (bf16_t)w[k * 512 + n];
}

// ---------- LayerNorm + transpose: x[b][c][s] f32 -> tn[(b*1024+s)][c] bf16
__global__ __launch_bounds__(256) void ln_kernel(const float* __restrict__ x,
                                                 const float* __restrict__ g,
                                                 const float* __restrict__ bta,
                                                 bf16_t* __restrict__ tn) {
  int wave = threadIdx.x >> 5, lane = threadIdx.x & 31;
  int row = blockIdx.x * 8 + wave;            // 0..8191  (b,s)
  int b = row >> 10, s = row & 1023;
  const float* xp = x + (size_t)(b * 512) * 1024 + s;
  float v[16]; float sum = 0.f, sq = 0.f;
#pragma unroll
  for (int i = 0; i < 16; ++i) {
    v[i] = xp[(size_t)(lane + i * 32) * 1024];
    sum += v[i]; sq += v[i] * v[i];
  }
#pragma unroll
  for (int m = 1; m < 32; m <<= 1) {
    sum += __shfl_xor(sum, m, 32);
    sq  += __shfl_xor(sq,  m, 32);
  }
  float mu   = sum * (1.f / 512.f);
  float var  = sq  * (1.f / 512.f) - mu * mu;
  float rstd = rsqrtf(var + 1e-5f);
  bf16_t* op = tn + (size_t)row * 512;
#pragma unroll
  for (int i = 0; i < 16; ++i) {
    int c = lane + i * 32;
    op[c] = (bf16_t)((v[i] - mu) * rstd * g[c] + bta[c]);
  }
}

// ---------- QKV projection: tn[8192x512] @ wt^T -> Q/K (B,H,S,D) or V^T (B,H,D,S)
// mode 0: Q (scaled by 1/sqrt(64)), mode 1: K, mode 2: V stored transposed
__global__ __launch_bounds__(256) void qkv_kernel(const bf16_t* __restrict__ tn,
                                                  const bf16_t* __restrict__ wt,
                                                  const float* __restrict__ bias,
                                                  bf16_t* __restrict__ dst,
                                                  int mode) {
  int wave = threadIdx.x >> 5, lane = threadIdx.x & 31;
  int job = blockIdx.x * 8 + wave;            // 0..4095
  int mtile = job >> 3;                       // 0..511  (16-row tile)
  int n64 = job & 7;                          // head index (64-col tile)
  int lg = lane >> 4, ln = lane & 15;

  v8f acc[4] = {};
  const bf16_t* arow = tn + (size_t)(mtile * 16 + ln) * 512 + lg * 16;
  const bf16_t* bcol0 = wt + (size_t)(n64 * 64 +  0 + ln) * 512 + lg * 16;
  const bf16_t* bcol1 = wt + (size_t)(n64 * 64 + 16 + ln) * 512 + lg * 16;
  const bf16_t* bcol2 = wt + (size_t)(n64 * 64 + 32 + ln) * 512 + lg * 16;
  const bf16_t* bcol3 = wt + (size_t)(n64 * 64 + 48 + ln) * 512 + lg * 16;

  // software-pipelined k-loop: prefetch next fragments before current WMMAs
  Frag16 aC, b0, b1, b2, b3;
  loadA(aC, arow, 0);
  loadB(b0, bcol0, 0); loadB(b1, bcol1, 0); loadB(b2, bcol2, 0); loadB(b3, bcol3, 0);
#pragma unroll
  for (int kt = 0; kt < 16; ++kt) {
    Frag16 aN, n0, n1, n2, n3;
    if (kt < 15) {
      int k0 = (kt + 1) * 32;
      loadA(aN, arow, k0);
      loadB(n0, bcol0, k0); loadB(n1, bcol1, k0);
      loadB(n2, bcol2, k0); loadB(n3, bcol3, k0);
    }
    acc[0] = wmma_bf16(aC.v, b0.v, acc[0]);
    acc[1] = wmma_bf16(aC.v, b1.v, acc[1]);
    acc[2] = wmma_bf16(aC.v, b2.v, acc[2]);
    acc[3] = wmma_bf16(aC.v, b3.v, acc[3]);
    if (kt < 15) { aC = aN; b0 = n0; b1 = n1; b2 = n2; b3 = n3; }
  }
#pragma unroll
  for (int sub = 0; sub < 4; ++sub) {
    int n = n64 * 64 + sub * 16 + ln;
    float bv = bias[n];
    int d = n & 63;
#pragma unroll
    for (int r = 0; r < 8; ++r) {
      int m = mtile * 16 + r + lg * 8;
      int b = m >> 10, s = m & 1023;
      float val = acc[sub][r] + bv;
      if (mode == 0) val *= 0.125f;
      size_t off;
      if (mode == 2) off = ((size_t)((b * 8 + n64) * 64 + d)) * 1024 + s;   // [B,H,D,S]
      else           off = ((size_t)((b * 8 + n64) * 1024 + s)) * 64 + d;   // [B,H,S,D]
      dst[off] = (bf16_t)val;
    }
  }
}

// ---------- flash attention: per (b,h), 16 query rows per wave, online softmax.
// K/V tiles staged block-wide in LDS via double-buffered async copies (ASYNCcnt).
__global__ __launch_bounds__(256) void attn_kernel(const bf16_t* __restrict__ Q,
                                                   const bf16_t* __restrict__ K,
                                                   const bf16_t* __restrict__ Vt,
                                                   bf16_t* __restrict__ O) {
  __shared__ __align__(16) bf16_t kbuf[2][32 * 64];   // K tile: 32 keys x 64 d
  __shared__ __align__(16) bf16_t vbuf[2][64 * 32];   // V^T tile: 64 d x 32 keys
  __shared__ __align__(16) bf16_t pslab[8][16 * 32];  // per-wave P tile
  int tid = threadIdx.x;
  int wave = tid >> 5, lane = tid & 31;
  int lg = lane >> 4, ln = lane & 15;
  int bh = blockIdx.x >> 3, qb = blockIdx.x & 7;
  int b = bh >> 3, h = bh & 7;
  const bf16_t* Qp = Q  + (size_t)bh * 65536;
  const bf16_t* Kp = K  + (size_t)bh * 65536;
  const bf16_t* Vp = Vt + (size_t)bh * 65536;
  int qtile = qb * 128 + wave * 16;

  // per-thread 16B async-copy chunks: K = 32 rows x 8 chunks, V = 64 rows x 4 chunks
  int krow = tid >> 3, kseg = tid & 7;
  int vrow = tid >> 2, vseg = tid & 3;
  auto issue_copy = [&](int buf, int j0) {
    unsigned long long gk = (unsigned long long)(Kp + (size_t)(j0 + krow) * 64 + kseg * 8);
    unsigned lk = (unsigned)(size_t)(&kbuf[buf][krow * 64 + kseg * 8]);
    asm volatile("global_load_async_to_lds_b128 %0, %1, off" :: "v"(lk), "v"(gk) : "memory");
    unsigned long long gv = (unsigned long long)(Vp + (size_t)vrow * 1024 + j0 + vseg * 8);
    unsigned lv = (unsigned)(size_t)(&vbuf[buf][vrow * 32 + vseg * 8]);
    asm volatile("global_load_async_to_lds_b128 %0, %1, off" :: "v"(lv), "v"(gv) : "memory");
  };

  Frag16 aq0, aq1;   // Q tile 16x64 as two A-frags (d 0..31, 32..63)
  {
    const bf16_t* qrow = Qp + (size_t)(qtile + ln) * 64 + lg * 16;
    aq0.q[0] = *reinterpret_cast<const v4u*>(qrow);
    aq0.q[1] = *reinterpret_cast<const v4u*>(qrow + 8);
    aq1.q[0] = *reinterpret_cast<const v4u*>(qrow + 32);
    aq1.q[1] = *reinterpret_cast<const v4u*>(qrow + 40);
  }

  float mrow[8], lrow[8];
#pragma unroll
  for (int r = 0; r < 8; ++r) { mrow[r] = -3.0e38f; lrow[r] = 0.f; }
  v8f accO[4] = {};
  bf16_t* slab = pslab[wave];

  issue_copy(0, 0);
  asm volatile("s_wait_asynccnt 0x0" ::: "memory");
  __syncthreads();

  for (int jt = 0; jt < 32; ++jt) {
    int cur = jt & 1;
    if (jt + 1 < 32) issue_copy(cur ^ 1, (jt + 1) * 32);

    v8f sc[2];
#pragma unroll
    for (int jh = 0; jh < 2; ++jh) {          // scores for 16 key columns (from LDS)
      const bf16_t* kb = &kbuf[cur][(jh * 16 + ln) * 64 + lg * 16];
      Frag16 k0, k1;
      k0.q[0] = *reinterpret_cast<const v4u*>(kb);
      k0.q[1] = *reinterpret_cast<const v4u*>(kb + 8);
      k1.q[0] = *reinterpret_cast<const v4u*>(kb + 32);
      k1.q[1] = *reinterpret_cast<const v4u*>(kb + 40);
      v8f s0 = {};
      s0 = wmma_bf16(aq0.v, k0.v, s0);
      s0 = wmma_bf16(aq1.v, k1.v, s0);
      sc[jh] = s0;
    }
    // online softmax over the 32 new columns (row = r + 8*lg, col = ln in group)
#pragma unroll
    for (int r = 0; r < 8; ++r) {
      float t = fmaxf(sc[0][r], sc[1][r]);
      t = fmaxf(t, __shfl_xor(t, 1, 32));
      t = fmaxf(t, __shfl_xor(t, 2, 32));
      t = fmaxf(t, __shfl_xor(t, 4, 32));
      t = fmaxf(t, __shfl_xor(t, 8, 32));
      float nm  = fmaxf(mrow[r], t);
      float sca = __expf(mrow[r] - nm);
      float p0  = __expf(sc[0][r] - nm);
      float p1  = __expf(sc[1][r] - nm);
      sc[0][r] = p0; sc[1][r] = p1;
      float rs = p0 + p1;
      rs += __shfl_xor(rs, 1, 32);
      rs += __shfl_xor(rs, 2, 32);
      rs += __shfl_xor(rs, 4, 32);
      rs += __shfl_xor(rs, 8, 32);
      lrow[r] = lrow[r] * sca + rs;
      mrow[r] = nm;
      accO[0][r] *= sca; accO[1][r] *= sca; accO[2][r] *= sca; accO[3][r] *= sca;
    }
    // D-layout -> A-layout via wave-private LDS slab
#pragma unroll
    for (int r = 0; r < 8; ++r) {
      int row = r + lg * 8;
      slab[row * 32 + ln]      = (bf16_t)sc[0][r];
      slab[row * 32 + 16 + ln] = (bf16_t)sc[1][r];
    }
    asm volatile("s_wait_dscnt 0x0" ::: "memory");
    Frag16 ap;
    ap.q[0] = *reinterpret_cast<const v4u*>(slab + ln * 32 + lg * 16);
    ap.q[1] = *reinterpret_cast<const v4u*>(slab + ln * 32 + lg * 16 + 8);
    // accO(16x64) += P(16x32) * V(32x64), V^T tile in LDS [d][32]
#pragma unroll
    for (int sub = 0; sub < 4; ++sub) {
      const bf16_t* vb = &vbuf[cur][(sub * 16 + ln) * 32 + lg * 16];
      Frag16 bv;
      bv.q[0] = *reinterpret_cast<const v4u*>(vb);
      bv.q[1] = *reinterpret_cast<const v4u*>(vb + 8);
      accO[sub] = wmma_bf16(ap.v, bv.v, accO[sub]);
    }
    asm volatile("s_wait_asynccnt 0x0" ::: "memory");
    __syncthreads();
  }
  // normalize + store O[b, s, h*64 + d] bf16
#pragma unroll
  for (int sub = 0; sub < 4; ++sub) {
#pragma unroll
    for (int r = 0; r < 8; ++r) {
      int srow = qtile + r + lg * 8;
      int d = sub * 16 + ln;
      float val = accO[sub][r] / lrow[r];
      O[((size_t)b * 1024 + srow) * 512 + h * 64 + d] = (bf16_t)val;
    }
  }
}

// ---------- output projection + bias + residual (residual read from x[b][c][s])
__global__ __launch_bounds__(256) void proj_kernel(const bf16_t* __restrict__ Om,
                                                   const bf16_t* __restrict__ wt,
                                                   const float* __restrict__ bo,
                                                   const float* __restrict__ x,
                                                   float* __restrict__ out) {
  int wave = threadIdx.x >> 5, lane = threadIdx.x & 31;
  int job = blockIdx.x * 8 + wave;            // 0..4095
  int mtile = job >> 3;
  int n64 = job & 7;
  int lg = lane >> 4, ln = lane & 15;

  v8f acc[4] = {};
  const bf16_t* arow = Om + (size_t)(mtile * 16 + ln) * 512 + lg * 16;
  const bf16_t* bcol0 = wt + (size_t)(n64 * 64 +  0 + ln) * 512 + lg * 16;
  const bf16_t* bcol1 = wt + (size_t)(n64 * 64 + 16 + ln) * 512 + lg * 16;
  const bf16_t* bcol2 = wt + (size_t)(n64 * 64 + 32 + ln) * 512 + lg * 16;
  const bf16_t* bcol3 = wt + (size_t)(n64 * 64 + 48 + ln) * 512 + lg * 16;

  Frag16 aC, b0, b1, b2, b3;
  loadA(aC, arow, 0);
  loadB(b0, bcol0, 0); loadB(b1, bcol1, 0); loadB(b2, bcol2, 0); loadB(b3, bcol3, 0);
#pragma unroll
  for (int kt = 0; kt < 16; ++kt) {
    Frag16 aN, n0, n1, n2, n3;
    if (kt < 15) {
      int k0 = (kt + 1) * 32;
      loadA(aN, arow, k0);
      loadB(n0, bcol0, k0); loadB(n1, bcol1, k0);
      loadB(n2, bcol2, k0); loadB(n3, bcol3, k0);
    }
    acc[0] = wmma_bf16(aC.v, b0.v, acc[0]);
    acc[1] = wmma_bf16(aC.v, b1.v, acc[1]);
    acc[2] = wmma_bf16(aC.v, b2.v, acc[2]);
    acc[3] = wmma_bf16(aC.v, b3.v, acc[3]);
    if (kt < 15) { aC = aN; b0 = n0; b1 = n1; b2 = n2; b3 = n3; }
  }
#pragma unroll
  for (int sub = 0; sub < 4; ++sub) {
    int n = n64 * 64 + sub * 16 + ln;
    float bb = bo[n];
#pragma unroll
    for (int r = 0; r < 8; ++r) {
      int m = mtile * 16 + r + lg * 8;
      int b = m >> 10, s = m & 1023;
      float res = x[((size_t)(b * 512 + n)) * 1024 + s];
      out[(size_t)m * 512 + n] = acc[sub][r] + bb + res;
    }
  }
}

extern "C" void kernel_launch(void* const* d_in, const int* in_sizes, int n_in,
                              void* d_out, int out_size, void* d_ws, size_t ws_size,
                              hipStream_t stream) {
  const float* x   = (const float*)d_in[0];
  const float* wq  = (const float*)d_in[1];
  const float* bq  = (const float*)d_in[2];
  const float* wk  = (const float*)d_in[3];
  const float* bk  = (const float*)d_in[4];
  const float* wv  = (const float*)d_in[5];
  const float* bv  = (const float*)d_in[6];
  const float* lng = (const float*)d_in[7];
  const float* lnb = (const float*)d_in[8];
  const float* wo  = (const float*)d_in[9];
  const float* bo  = (const float*)d_in[10];
  float* out = (float*)d_out;

  // bf16 workspace layout (elements): tn | Q | K | V^T | O | wqT wkT wvT woT
  bf16_t* ws  = (bf16_t*)d_ws;
  bf16_t* tn  = ws;                        // 8192*512
  bf16_t* q   = ws + 1 * 4194304;          // [B,H,S,D]
  bf16_t* k   = ws + 2 * 4194304;          // [B,H,S,D]
  bf16_t* vt  = ws + 3 * 4194304;          // [B,H,D,S]
  bf16_t* o   = ws + 4 * 4194304;          // [B*S, 512]
  bf16_t* wqT = ws + 5 * 4194304;
  bf16_t* wkT = wqT + 262144;
  bf16_t* wvT = wkT + 262144;
  bf16_t* woT = wvT + 262144;

  wt_kernel<<<1024, 256, 0, stream>>>(wq, wqT);
  wt_kernel<<<1024, 256, 0, stream>>>(wk, wkT);
  wt_kernel<<<1024, 256, 0, stream>>>(wv, wvT);
  wt_kernel<<<1024, 256, 0, stream>>>(wo, woT);
  ln_kernel<<<1024, 256, 0, stream>>>(x, lng, lnb, tn);
  qkv_kernel<<<512, 256, 0, stream>>>(tn, wqT, bq, q, 0);
  qkv_kernel<<<512, 256, 0, stream>>>(tn, wkT, bk, k, 1);
  qkv_kernel<<<512, 256, 0, stream>>>(tn, wvT, bv, vt, 2);
  attn_kernel<<<512, 256, 0, stream>>>(q, k, vt, o);
  proj_kernel<<<512, 256, 0, stream>>>(o, woT, bo, x, out);
}
